// TeacherNetwork_77232101916761
// MI455X (gfx1250) — matrix operand order repaired and checked by reading, not compiled, not run on hardware
//
#include <hip/hip_runtime.h>
#include <hip/hip_bf16.h>

#define NP 16384
#define DM 512
#define KNB 16

typedef __attribute__((ext_vector_type(16))) __bf16 v16bf;
typedef __attribute__((ext_vector_type(8)))  __bf16 v8bf;
typedef __attribute__((ext_vector_type(8)))  float  v8f;

// ---------------------------------------------------------------------------
// Prep: convert weights to bf16, combined bias, zero-init f0 (f32 + bf16).
// ---------------------------------------------------------------------------
__global__ void prep_kernel(const float* __restrict__ w1, const float* __restrict__ w2,
                            const float* __restrict__ wr, const float* __restrict__ b2,
                            const float* __restrict__ br,
                            __bf16* __restrict__ w1b, __bf16* __restrict__ w2b,
                            __bf16* __restrict__ wrb, float* __restrict__ biasc,
                            float* __restrict__ f0f, __bf16* __restrict__ f0b)
{
    long t = (long)blockIdx.x * blockDim.x + threadIdx.x;
    const long n1 = 128L * 512;              // w_mlp1
    const long n2 = n1 + 512L * 512;         // w_mlp2
    const long n3 = n2 + 512L * 512;         // w_res
    const long n4 = n3 + 512;                // bias_c = b_mlp2 + b_res
    const long n5 = n4 + (long)NP * DM;      // f0 zero-init
    if (t < n1) {
        w1b[t] = (__bf16)w1[t];
    } else if (t < n2) {
        long i = t - n1; w2b[i] = (__bf16)w2[i];
    } else if (t < n3) {
        long i = t - n2; wrb[i] = (__bf16)wr[i];
    } else if (t < n4) {
        long i = t - n3; biasc[i] = b2[i] + br[i];
    } else if (t < n5) {
        long i = t - n4; f0f[i] = 0.0f; f0b[i] = (__bf16)0.0f;
    }
}

// ---------------------------------------------------------------------------
// geo_mean[n, 0:3] = mean_k (p_n - p_knn[n,k]);  geo_mean[n,3] = mean_k ||diff||
// ---------------------------------------------------------------------------
__global__ void geo_mean_kernel(const float* __restrict__ pts,
                                const long long* __restrict__ knn,
                                float* __restrict__ gm)
{
    int n = blockIdx.x * blockDim.x + threadIdx.x;
    if (n >= NP) return;
    float px = pts[3 * n + 0], py = pts[3 * n + 1], pz = pts[3 * n + 2];
    float sx = 0.f, sy = 0.f, sz = 0.f, sn = 0.f;
#pragma unroll
    for (int k = 0; k < KNB; ++k) {
        long long idx = knn[(long)n * KNB + k];
        float dx = px - pts[3 * idx + 0];
        float dy = py - pts[3 * idx + 1];
        float dz = pz - pts[3 * idx + 2];
        sx += dx; sy += dy; sz += dz;
        sn += sqrtf(dx * dx + dy * dy + dz * dz);
    }
    const float inv = 1.0f / (float)KNB;
    gm[4 * n + 0] = sx * inv;
    gm[4 * n + 1] = sy * inv;
    gm[4 * n + 2] = sz * inv;
    gm[4 * n + 3] = sn * inv;
}

// ---------------------------------------------------------------------------
// Constant (across blocks) geometric halves of the LFA concats:
//   x2[:, 0:128] = geo_mean @ w_lfa1.T + b_lfa1
//   x3[:, 0:256] = geo_mean @ w_lfa2.T + b_lfa2
// ---------------------------------------------------------------------------
__global__ void gfeat_kernel(const float* __restrict__ gm,
                             const float* __restrict__ wl1, const float* __restrict__ bl1,
                             const float* __restrict__ wl2, const float* __restrict__ bl2,
                             __bf16* __restrict__ x2, __bf16* __restrict__ x3)
{
    long t = (long)blockIdx.x * blockDim.x + threadIdx.x;
    if (t >= (long)NP * 384) return;
    int n = (int)(t / 384);
    int c = (int)(t % 384);
    const float* g = gm + 4 * n;
    if (c < 128) {
        const float* w = wl1 + 4 * c;
        float v = g[0]*w[0] + g[1]*w[1] + g[2]*w[2] + g[3]*w[3] + bl1[c];
        x2[(long)n * 256 + c] = (__bf16)v;
    } else {
        int c2 = c - 128;
        const float* w = wl2 + 4 * c2;
        float v = g[0]*w[0] + g[1]*w[1] + g[2]*w[2] + g[3]*w[3] + bl2[c2];
        x3[(long)n * 512 + c2] = (__bf16)v;
    }
}

// ---------------------------------------------------------------------------
// WMMA fragment loads per CDNA5 16-bit layouts.
// A 16x32 (MxK): lane = m + 16h; elems 0..7 -> K = 8h+j, elems 8..15 -> K = 16+8h+(j-8)
// B 32x16 (KxN): lane = n + 16h; elems 0..7 -> K = 8h+j, elems 8..15 -> K = 16+8h+(j-8)
// ---------------------------------------------------------------------------
__device__ inline v16bf load_a_frag(const __bf16* __restrict__ row, int k0, int h)
{
    v8bf lo = *(const v8bf*)(row + k0 + 8 * h);
    v8bf hi = *(const v8bf*)(row + k0 + 16 + 8 * h);
    return __builtin_shufflevector(lo, hi, 0,1,2,3,4,5,6,7,8,9,10,11,12,13,14,15);
}
__device__ inline v16bf load_b_frag(const __bf16* __restrict__ row, int k0, int h)
{
    v8bf lo = *(const v8bf*)(row + k0 + 8 * h);
    v8bf hi = *(const v8bf*)(row + k0 + 16 + 8 * h);
    return __builtin_shufflevector(lo, hi, 0,1,2,3,4,5,6,7,8,9,10,11,12,13,14,15);
}
__device__ inline v8f wmma_bf16(v16bf a, v16bf b, v8f c)
{
    return __builtin_amdgcn_wmma_f32_16x16x32_bf16(false, a, false, b,
                                                   (short)0, c, false, false);
}

// ---------------------------------------------------------------------------
// GEMM 1: x1[NP,128] = bf16( f0_bf16[NP,512] @ w1b[128,512].T + b1 )
// One wave per 32x32 output block (2x2 tiles): 2 loads per WMMA.
// ---------------------------------------------------------------------------
__global__ void gemm_x1_kernel(const __bf16* __restrict__ A,   // [NP, 512]
                               const __bf16* __restrict__ W,   // [128, 512]
                               const float*  __restrict__ bias,// [128]
                               __bf16* __restrict__ Out)       // [NP, 128]
{
    const int DIN = 512, DOUT = 128, NB = DOUT / 32;   // 4 n-blocks
    int wave = blockIdx.x * (blockDim.x >> 5) + (threadIdx.x >> 5);
    int mB = wave / NB, nB = wave % NB;
    int lane = threadIdx.x & 31;
    int h = lane >> 4, r = lane & 15;
    const __bf16* arow0 = A + (size_t)(mB * 32 + r) * DIN;
    const __bf16* arow1 = A + (size_t)(mB * 32 + 16 + r) * DIN;
    const __bf16* wrow0 = W + (size_t)(nB * 32 + r) * DIN;
    const __bf16* wrow1 = W + (size_t)(nB * 32 + 16 + r) * DIN;
    v8f acc00 = {}, acc01 = {}, acc10 = {}, acc11 = {};
#pragma unroll
    for (int k0 = 0; k0 < DIN; k0 += 32) {
        v16bf a0 = load_a_frag(arow0, k0, h);
        v16bf a1 = load_a_frag(arow1, k0, h);
        v16bf b0 = load_b_frag(wrow0, k0, h);
        v16bf b1 = load_b_frag(wrow1, k0, h);
        acc00 = wmma_bf16(a0, b0, acc00);
        acc01 = wmma_bf16(a0, b1, acc01);
        acc10 = wmma_bf16(a1, b0, acc10);
        acc11 = wmma_bf16(a1, b1, acc11);
    }
    int col0 = nB * 32 + r, col1 = col0 + 16;
    float bv0 = bias[col0], bv1 = bias[col1];
#pragma unroll
    for (int e = 0; e < 8; ++e) {
        int row0 = mB * 32 + e + 8 * h;
        int row1 = row0 + 16;
        Out[(size_t)row0 * DOUT + col0] = (__bf16)(acc00[e] + bv0);
        Out[(size_t)row0 * DOUT + col1] = (__bf16)(acc01[e] + bv1);
        Out[(size_t)row1 * DOUT + col0] = (__bf16)(acc10[e] + bv0);
        Out[(size_t)row1 * DOUT + col1] = (__bf16)(acc11[e] + bv1);
    }
}

// ---------------------------------------------------------------------------
// Final fused GEMM per block:
//   f0_out = x3 @ w2b.T + f0_bf16 @ wrb.T + f0_f32 + (b_mlp2 + b_res)
// One wave per 32x32 output block; both GEMMs accumulate into the same tiles.
// Writes f32 (residual precision / final output) and bf16 mirrors.
// ---------------------------------------------------------------------------
__global__ void gemm_out_kernel(const __bf16* __restrict__ X3,  // [NP,512]
                                const __bf16* __restrict__ W2,  // [512,512]
                                const __bf16* __restrict__ F0b, // [NP,512]
                                const __bf16* __restrict__ WR,  // [512,512]
                                const float*  __restrict__ F0f, // [NP,512]
                                const float*  __restrict__ biasc,
                                float*  __restrict__ OutF,
                                __bf16* __restrict__ OutB)
{
    const int DIN = 512, DOUT = 512, NB = DOUT / 32;   // 16 n-blocks
    int wave = blockIdx.x * (blockDim.x >> 5) + (threadIdx.x >> 5);
    int mB = wave / NB, nB = wave % NB;
    int lane = threadIdx.x & 31;
    int h = lane >> 4, r = lane & 15;
    const __bf16* xrow0 = X3  + (size_t)(mB * 32 + r) * DIN;
    const __bf16* xrow1 = X3  + (size_t)(mB * 32 + 16 + r) * DIN;
    const __bf16* frow0 = F0b + (size_t)(mB * 32 + r) * DIN;
    const __bf16* frow1 = F0b + (size_t)(mB * 32 + 16 + r) * DIN;
    const __bf16* w2r0  = W2  + (size_t)(nB * 32 + r) * DIN;
    const __bf16* w2r1  = W2  + (size_t)(nB * 32 + 16 + r) * DIN;
    const __bf16* wrr0  = WR  + (size_t)(nB * 32 + r) * DIN;
    const __bf16* wrr1  = WR  + (size_t)(nB * 32 + 16 + r) * DIN;
    v8f acc00 = {}, acc01 = {}, acc10 = {}, acc11 = {};
#pragma unroll
    for (int k0 = 0; k0 < DIN; k0 += 32) {
        v16bf ax0 = load_a_frag(xrow0, k0, h);
        v16bf ax1 = load_a_frag(xrow1, k0, h);
        v16bf bw0 = load_b_frag(w2r0, k0, h);
        v16bf bw1 = load_b_frag(w2r1, k0, h);
        acc00 = wmma_bf16(ax0, bw0, acc00);
        acc01 = wmma_bf16(ax0, bw1, acc01);
        acc10 = wmma_bf16(ax1, bw0, acc10);
        acc11 = wmma_bf16(ax1, bw1, acc11);
        v16bf af0 = load_a_frag(frow0, k0, h);
        v16bf af1 = load_a_frag(frow1, k0, h);
        v16bf br0 = load_b_frag(wrr0, k0, h);
        v16bf br1 = load_b_frag(wrr1, k0, h);
        acc00 = wmma_bf16(af0, br0, acc00);
        acc01 = wmma_bf16(af0, br1, acc01);
        acc10 = wmma_bf16(af1, br0, acc10);
        acc11 = wmma_bf16(af1, br1, acc11);
    }
    int col0 = nB * 32 + r, col1 = col0 + 16;
    float bv0 = biasc[col0], bv1 = biasc[col1];
#pragma unroll
    for (int e = 0; e < 8; ++e) {
        int row0 = mB * 32 + e + 8 * h;
        int row1 = row0 + 16;
        float v00 = acc00[e] + F0f[(size_t)row0 * DOUT + col0] + bv0;
        float v01 = acc01[e] + F0f[(size_t)row0 * DOUT + col1] + bv1;
        float v10 = acc10[e] + F0f[(size_t)row1 * DOUT + col0] + bv0;
        float v11 = acc11[e] + F0f[(size_t)row1 * DOUT + col1] + bv1;
        OutF[(size_t)row0 * DOUT + col0] = v00;
        OutF[(size_t)row0 * DOUT + col1] = v01;
        OutF[(size_t)row1 * DOUT + col0] = v10;
        OutF[(size_t)row1 * DOUT + col1] = v11;
        OutB[(size_t)row0 * DOUT + col0] = (__bf16)v00;
        OutB[(size_t)row0 * DOUT + col1] = (__bf16)v01;
        OutB[(size_t)row1 * DOUT + col0] = (__bf16)v10;
        OutB[(size_t)row1 * DOUT + col1] = (__bf16)v11;
    }
}

// ---------------------------------------------------------------------------
// Neighbor average: Out[n, coff + c] = (1/16) sum_k In[knn[n,k], c]
// One wave per point; lanes stride columns (NC = cols/32 per lane).
// ---------------------------------------------------------------------------
template <int NC>
__global__ void gather_avg_kernel(const long long* __restrict__ knn,
                                  const __bf16* __restrict__ In,  // [NP, NC*32]
                                  __bf16* __restrict__ Out, int ldo, int coff)
{
    int n = blockIdx.x * (blockDim.x >> 5) + (threadIdx.x >> 5);
    if (n >= NP) return;
    int lane = threadIdx.x & 31;
    float acc[NC];
#pragma unroll
    for (int j = 0; j < NC; ++j) acc[j] = 0.0f;
#pragma unroll
    for (int k = 0; k < KNB; ++k) {
        long long idx = knn[(long)n * KNB + k];
        const __bf16* rowp = In + (size_t)idx * (NC * 32);
#pragma unroll
        for (int j = 0; j < NC; ++j)
            acc[j] += (float)rowp[lane + 32 * j];
    }
    const float inv = 1.0f / (float)KNB;
#pragma unroll
    for (int j = 0; j < NC; ++j)
        Out[(size_t)n * ldo + coff + lane + 32 * j] = (__bf16)(acc[j] * inv);
}

// ---------------------------------------------------------------------------
extern "C" void kernel_launch(void* const* d_in, const int* in_sizes, int n_in,
                              void* d_out, int out_size, void* d_ws, size_t ws_size,
                              hipStream_t stream)
{
    (void)in_sizes; (void)n_in; (void)out_size; (void)ws_size;

    const float*     inputs = (const float*)d_in[0];
    const long long* knn    = (const long long*)d_in[1];
    const float* w_mlp1 = (const float*)d_in[2];
    const float* b_mlp1 = (const float*)d_in[3];
    const float* w_lfa1 = (const float*)d_in[4];
    const float* b_lfa1 = (const float*)d_in[5];
    const float* w_lfa2 = (const float*)d_in[6];
    const float* b_lfa2 = (const float*)d_in[7];
    const float* w_mlp2 = (const float*)d_in[8];
    const float* b_mlp2 = (const float*)d_in[9];
    const float* w_res  = (const float*)d_in[10];
    const float* b_res  = (const float*)d_in[11];

    // Workspace carve-up (all chunks 256B-aligned; ~98 MB total).
    char* ws = (char*)d_ws;
    size_t off = 0;
    auto carve = [&](size_t bytes) -> char* {
        char* p = ws + off;
        off += (bytes + 255) & ~(size_t)255;
        return p;
    };
    float*  geo   = (float*) carve((size_t)NP * 4 * 4);
    __bf16* x1    = (__bf16*)carve((size_t)NP * 128 * 2);
    __bf16* x2    = (__bf16*)carve((size_t)NP * 256 * 2);
    __bf16* x3    = (__bf16*)carve((size_t)NP * 512 * 2);
    float*  f0b_f = (float*) carve((size_t)NP * 512 * 4);
    __bf16* f0a_b = (__bf16*)carve((size_t)NP * 512 * 2);
    __bf16* f0b_b = (__bf16*)carve((size_t)NP * 512 * 2);
    __bf16* w1b   = (__bf16*)carve(128 * 512 * 2);
    __bf16* w2b   = (__bf16*)carve(512 * 512 * 2);
    __bf16* wrb   = (__bf16*)carve(512 * 512 * 2);
    float*  biasc = (float*) carve(512 * 4);
    float*  f0a_f = (float*)d_out;   // ping buffer A (f32) IS the output

    // --- prep: weight conversion + bias sum + zero f0 (ping A) ---
    {
        long total = 128L*512 + 512L*512 + 512L*512 + 512 + (long)NP * DM;
        int blocks = (int)((total + 255) / 256);
        prep_kernel<<<blocks, 256, 0, stream>>>(w_mlp1, w_mlp2, w_res, b_mlp2, b_res,
                                                w1b, w2b, wrb, biasc, f0a_f, f0a_b);
    }
    // --- geo_mean ---
    geo_mean_kernel<<<NP / 256, 256, 0, stream>>>(inputs, knn, geo);
    // --- constant geometric halves of x2 / x3 ---
    {
        long total = (long)NP * 384;
        gfeat_kernel<<<(int)((total + 255) / 256), 256, 0, stream>>>(
            geo, w_lfa1, b_lfa1, w_lfa2, b_lfa2, x2, x3);
    }

    // --- 4 residual blocks, ping-pong f0 between (A = d_out, B = scratch) ---
    const int gemm1_blocks = (NP / 32) * (128 / 32) / 4;   // 2x2-tile waves, 4/block
    const int gemm2_blocks = (NP / 32) * (512 / 32) / 4;
    const int gath_blocks  = NP / 8;                       // 8 waves/block of 256 thr

    for (int blk = 0; blk < 4; ++blk) {
        const bool even = (blk % 2) == 0;
        const float*  fin_f  = even ? f0a_f : f0b_f;
        const __bf16* fin_b  = even ? f0a_b : f0b_b;
        float*        fout_f = even ? f0b_f : f0a_f;
        __bf16*       fout_b = even ? f0b_b : f0a_b;

        gemm_x1_kernel<<<gemm1_blocks, 128, 0, stream>>>(fin_b, w1b, b_mlp1, x1);
        gather_avg_kernel<4><<<gath_blocks, 256, 0, stream>>>(knn, x1, x2, 256, 128);
        gather_avg_kernel<8><<<gath_blocks, 256, 0, stream>>>(knn, x2, x3, 512, 256);
        gemm_out_kernel<<<gemm2_blocks, 128, 0, stream>>>(x3, w2b, fin_b, wrb,
                                                          fin_f, biasc, fout_f, fout_b);
    }
    // blk 3 (odd) wrote fout_f = f0a_f == d_out: final result already in place.
}
